// RAFTModel_86406152061535
// MI455X (gfx1250) — compile-verified, matrix operand order checked by reading, not run on hardware
//
#include <hip/hip_runtime.h>
#include <stdint.h>

// ============================================================================
// RAFT forward for MI455X (gfx1250, wave32, WMMA).
// All matmul-shaped work (convs via im2col, correlation volume) runs through
// v_wmma_f32_16x16x32_bf16. The corr GEMM stages B-tiles with the Tensor Data
// Mover (tensor_load_to_lds + s_wait_tensorcnt). Corr pyramid (~85MB) is kept
// resident (fits MI455X 192MB L2); lookup kernel prefetches via
// global_prefetch_b8. 256-thread blocks = 8 wave32 waves.
// ============================================================================

typedef __attribute__((ext_vector_type(16))) __bf16 v16bf;
typedef __attribute__((ext_vector_type(8)))  float  v8f;
typedef __attribute__((ext_vector_type(4)))  unsigned int u32x4;
typedef __attribute__((ext_vector_type(8)))  int    i32x8;
typedef __attribute__((ext_vector_type(4)))  int    i32x4;

static __device__ __forceinline__ __bf16 f2bf(float f) {
  unsigned u = __builtin_bit_cast(unsigned, f);
  unsigned r = (u + 0x7FFFu + ((u >> 16) & 1u)) >> 16;   // round-to-nearest-even
  unsigned short s = (unsigned short)r;
  return __builtin_bit_cast(__bf16, s);
}

// Flat pointer to __shared__ -> LDS byte offset (flat LDS addr low 32 bits are
// the in-LDS offset; aperture bits live in [63:32] per ISA 10.2).
static __device__ __forceinline__ unsigned lds_byte_offset(const void* p) {
  return (unsigned)(unsigned long long)(uintptr_t)p;
}

// ---------------------------------------------------------------------------
// f32 -> bf16 conversion
// ---------------------------------------------------------------------------
__global__ void k_f32_to_bf16(const float* __restrict__ s, __bf16* __restrict__ d, long n) {
  long i = (long)blockIdx.x * blockDim.x + threadIdx.x;
  if (i < n) d[i] = f2bf(s[i]);
}

// ---------------------------------------------------------------------------
// im2col: x (C,H,W) f32 -> col [M=OH*OW][K=C*KH*KW] bf16 (kw fastest, matches
// OIHW weight flattening)
// ---------------------------------------------------------------------------
__global__ void k_im2col(const float* __restrict__ x, __bf16* __restrict__ col,
                         int C, int H, int W, int KH, int KW, int SH, int SW,
                         int PH, int PW, int OH, int OW) {
  long K = (long)C * KH * KW;
  long M = (long)OH * OW;
  long total = M * K;
  long idx = (long)blockIdx.x * blockDim.x + threadIdx.x;
  if (idx >= total) return;
  long m = idx / K;
  long k = idx - m * K;
  int ow = (int)(m % OW), oh = (int)(m / OW);
  int kw = (int)(k % KW);
  long t = k / KW;
  int kh = (int)(t % KH);
  int c  = (int)(t / KH);
  int iy = oh * SH - PH + kh;
  int ix = ow * SW - PW + kw;
  float v = 0.0f;
  if (iy >= 0 && iy < H && ix >= 0 && ix < W) v = x[((size_t)c * H + iy) * W + ix];
  col[idx] = f2bf(v);
}

// ---------------------------------------------------------------------------
// WMMA bf16 GEMM:  C = act( scale * A*B + bias )
//   A: MxK (or KxM if transA), B: KxN (or NxK if transB), bf16
//   C: MxN f32 (or NxM column layout if transC -> NCHW conv outputs)
// Tile 128x128x32; 8 waves (4 row-groups x 2 col-groups); each wave owns a
// 32x64 sub-tile = 2x4 WMMA frags -> 8 WMMAs per K-step vs 6 LDS frag loads.
// use_tdm (valid when !transB): stage B tiles with TENSOR_LOAD_TO_LDS.
// ---------------------------------------------------------------------------
#define GEMM_TM 128
#define GEMM_TN 128
#define GEMM_TK 32

__global__ __launch_bounds__(256) void k_gemm(
    const __bf16* __restrict__ A, const __bf16* __restrict__ B,
    const float* __restrict__ bias, float* __restrict__ C,
    int M, int N, int K, int transA, int transB, int transC,
    int act, float scale, int use_tdm) {
  __shared__ __align__(16) __bf16 As[GEMM_TM][GEMM_TK];    // 8 KB
  __shared__ __align__(16) __bf16 Bs[GEMM_TN][GEMM_TK];    // 8 KB (N-major, K contiguous)
  __shared__ __align__(16) __bf16 Braw[GEMM_TK][GEMM_TN];  // 8 KB (TDM landing pad)

  const int tid  = threadIdx.x;
  const int lane = tid & 31;
  const int wave = tid >> 5;
  const int wr   = wave & 3;   // wave row group (4): rows wr*32..wr*32+31
  const int wc   = wave >> 2;  // wave col group (2): cols wc*64..wc*64+63
  const int half = lane >> 4;
  const int lm   = lane & 15;

  const int m0 = blockIdx.y * GEMM_TM;
  const int n0 = blockIdx.x * GEMM_TN;

  const unsigned short zb16 = 0;
  const __bf16 bz = __builtin_bit_cast(__bf16, zb16);

  v8f vz = {};
  v8f acc[2][4];
#pragma unroll
  for (int i = 0; i < 2; ++i)
#pragma unroll
    for (int j = 0; j < 4; ++j) acc[i][j] = vz;

  for (int k0 = 0; k0 < K; k0 += GEMM_TK) {
    // ---- stage A tile ----
    for (int i = tid; i < GEMM_TM * GEMM_TK; i += 256) {
      int r = i >> 5, c = i & 31;
      int gr = m0 + r, gc = k0 + c;
      __bf16 v = bz;
      if (gr < M && gc < K)
        v = transA ? A[(size_t)gc * M + gr] : A[(size_t)gr * K + gc];
      As[r][c] = v;
    }
    // ---- stage B tile ----
    if (use_tdm && !transB) {
      // Tensor Data Mover: 2D tile (TK rows x TN cols) of row-major KxN bf16.
      if (tid == 0) {
        unsigned ldsoff = lds_byte_offset(&Braw[0][0]);
        unsigned long long ga =
            (unsigned long long)(uintptr_t)(B + (size_t)k0 * N + n0);
        unsigned dim0 = (unsigned)(N - n0);   // row length remaining (elements)
        unsigned dim1 = (unsigned)(K - k0);   // rows remaining
        unsigned long long s0 = (unsigned long long)N;  // row stride (elements)
        u32x4 g0;
        g0.x = 1u;                    // count=1, no gather
        g0.y = ldsoff;                // lds_addr
        g0.z = (unsigned)ga;          // global_addr[31:0]
        g0.w = (unsigned)((ga >> 32) & 0x01FFFFFFull) | (2u << 30);  // addr hi | type=2
        i32x8 g1;
        g1[0] = (int)(1u << 16);                                   // data_size=1 (2B)
        g1[1] = (int)((dim0 & 0xFFFFu) << 16);                     // tensor_dim0 lo
        g1[2] = (int)((dim0 >> 16) | ((dim1 & 0xFFFFu) << 16));    // dim0 hi | dim1 lo
        g1[3] = (int)((dim1 >> 16) | ((unsigned)GEMM_TN << 16));   // dim1 hi | tile_dim0
        g1[4] = (int)(unsigned)GEMM_TK;                            // tile_dim1 (tile_dim2=0)
        g1[5] = (int)(unsigned)s0;                                 // dim0_stride lo32
        g1[6] = (int)((unsigned)(s0 >> 32) & 0xFFFFu);             // dim0_stride hi16
        g1[7] = 0;
        i32x4 gz4 = {0, 0, 0, 0};
        i32x8 gz8 = {0, 0, 0, 0, 0, 0, 0, 0};
        // 6-arg form (clang-23 / therock-10.0 headers): g0, g1, g2, g3, ext, cpol
        __builtin_amdgcn_tensor_load_to_lds(g0, g1, gz4, gz4, gz8, 0);
      }
      __builtin_amdgcn_s_wait_tensorcnt(0);
      __syncthreads();
      for (int i = tid; i < GEMM_TN * GEMM_TK; i += 256) {
        int n = i >> 5, c = i & 31;
        Bs[n][c] = Braw[c][n];  // transpose to N-major for contiguous B frags
      }
    } else {
      for (int i = tid; i < GEMM_TN * GEMM_TK; i += 256) {
        int n = i >> 5, c = i & 31;
        int gk = k0 + c, gn = n0 + n;
        __bf16 v = bz;
        if (gk < K && gn < N)
          v = transB ? B[(size_t)gn * K + gk] : B[(size_t)gk * N + gn];
        Bs[n][c] = v;
      }
    }
    __syncthreads();
    // ---- WMMA compute: each wave 2x4 frags of 16x16 ----
#pragma unroll
    for (int i = 0; i < 2; ++i) {
      int ar = wr * 32 + i * 16 + lm;
      v16bf a = *(const v16bf*)&As[ar][half * 16];
#pragma unroll
      for (int j = 0; j < 4; ++j) {
        int bn = wc * 64 + j * 16 + lm;
        v16bf b = *(const v16bf*)&Bs[bn][half * 16];
        acc[i][j] = __builtin_amdgcn_wmma_f32_16x16x32_bf16(
            false, a, false, b, (short)0, acc[i][j], false, false);
      }
    }
    __syncthreads();
  }

  // ---- epilogue: C/D layout -> VGPR r: M = r + 8*half, N = lm ----
#pragma unroll
  for (int i = 0; i < 2; ++i)
#pragma unroll
    for (int j = 0; j < 4; ++j) {
      int col = n0 + wc * 64 + j * 16 + lm;
#pragma unroll
      for (int r = 0; r < 8; ++r) {
        int row = m0 + wr * 32 + i * 16 + half * 8 + r;
        if (row < M && col < N) {
          float v = acc[i][j][r] * scale;
          if (bias) v += bias[col];
          if (act == 1) v = fmaxf(v, 0.0f);
          if (transC) C[(size_t)col * M + row] = v;
          else        C[(size_t)row * N + col] = v;
        }
      }
    }
}

// ---------------------------------------------------------------------------
// Instance norm (+relu), one block per channel
// ---------------------------------------------------------------------------
__global__ void k_instnorm_relu(float* __restrict__ x, int HW) {
  float* p = x + (size_t)blockIdx.x * HW;
  __shared__ float s1[256], s2[256];
  float a = 0.f, b = 0.f;
  for (int i = threadIdx.x; i < HW; i += 256) { float v = p[i]; a += v; b += v * v; }
  s1[threadIdx.x] = a; s2[threadIdx.x] = b;
  __syncthreads();
  for (int s = 128; s > 0; s >>= 1) {
    if (threadIdx.x < s) { s1[threadIdx.x] += s1[threadIdx.x + s]; s2[threadIdx.x] += s2[threadIdx.x + s]; }
    __syncthreads();
  }
  float mean = s1[0] / HW;
  float var  = s2[0] / HW - mean * mean;
  float rs = rsqrtf(var + 1e-5f);
  for (int i = threadIdx.x; i < HW; i += 256)
    p[i] = fmaxf((p[i] - mean) * rs, 0.0f);
}

// ---------------------------------------------------------------------------
// misc pointwise
// ---------------------------------------------------------------------------
__global__ void k_split_tanh_relu(const float* __restrict__ c, float* __restrict__ net,
                                  float* __restrict__ inp, int HW) {
  long n = 128l * HW;
  long i = (long)blockIdx.x * blockDim.x + threadIdx.x;
  if (i < n) { net[i] = tanhf(c[i]); inp[i] = fmaxf(c[n + i], 0.0f); }
}

__global__ void k_pool2x2(const float* __restrict__ in, float* __restrict__ out,
                          int n, int h, int w) {
  int oh = h >> 1, ow = w >> 1;
  long total = (long)n * oh * ow;
  long i = (long)blockIdx.x * blockDim.x + threadIdx.x;
  if (i >= total) return;
  int x = (int)(i % ow);
  long t = i / ow;
  int y = (int)(t % oh);
  long m = t / oh;
  const float* p = in + ((size_t)m * h + 2 * y) * w + 2 * x;
  out[i] = 0.25f * (p[0] + p[1] + p[w] + p[w + 1]);
}

__global__ void k_init_coords(float* __restrict__ c0, float* __restrict__ c1,
                              const float* __restrict__ fi, int h, int w) {
  int HW = h * w;
  long i = (long)blockIdx.x * blockDim.x + threadIdx.x;
  if (i >= 2l * HW) return;
  int ch = (int)(i / HW), p = (int)(i % HW);
  float base = (ch == 0) ? (float)(p % w) : (float)(p / w);
  c0[i] = base;
  c1[i] = base + fi[i];
}

__global__ void k_sub(const float* a, const float* b, float* o, long n) {
  long i = (long)blockIdx.x * blockDim.x + threadIdx.x;
  if (i < n) o[i] = a[i] - b[i];
}
__global__ void k_add_inplace(float* a, const float* b, long n) {
  long i = (long)blockIdx.x * blockDim.x + threadIdx.x;
  if (i < n) a[i] += b[i];
}
__global__ void k_sigmoid_mul(const float* zp, const float* rp, const float* h,
                              float* z, float* rh, long n) {
  long i = (long)blockIdx.x * blockDim.x + threadIdx.x;
  if (i >= n) return;
  float zz = 1.0f / (1.0f + expf(-zp[i]));
  float rr = 1.0f / (1.0f + expf(-rp[i]));
  z[i] = zz;
  rh[i] = rr * h[i];
}
__global__ void k_gru_out(const float* z, const float* qp, float* h, long n) {
  long i = (long)blockIdx.x * blockDim.x + threadIdx.x;
  if (i >= n) return;
  float zz = z[i];
  h[i] = (1.0f - zz) * h[i] + zz * tanhf(qp[i]);
}
__global__ void k_softplus(float* x, long n) {
  long i = (long)blockIdx.x * blockDim.x + threadIdx.x;
  if (i >= n) return;
  float v = x[i];
  x[i] = (v > 20.0f) ? v : log1pf(expf(v));
}

// ---------------------------------------------------------------------------
// Correlation pyramid lookup (bilinear, reference grid_sample semantics).
// out layout: [ch = lev*81+tap][pix]  (NCHW for the 1x1 conv that follows)
// ---------------------------------------------------------------------------
__global__ void k_corr_lookup(const float* __restrict__ l0, const float* __restrict__ l1,
                              const float* __restrict__ l2, const float* __restrict__ l3,
                              const float* __restrict__ coords, float* __restrict__ out,
                              int h, int w) {
  const int HW = h * w;
  long total = (long)HW * 4 * 81;
  long i = (long)blockIdx.x * blockDim.x + threadIdx.x;
  if (i >= total) return;
  int pix = (int)(i / (4 * 81));
  int rem = (int)(i % (4 * 81));
  int lev = rem / 81, tap = rem % 81;
  float d0 = (float)(tap / 9) - 4.0f;  // delta component 0 (grid x)
  float d1 = (float)(tap % 9) - 4.0f;  // delta component 1 (grid y)
  float cx = coords[pix];
  float cy = coords[HW + pix];
  const float* base = (lev == 0) ? l0 : (lev == 1) ? l1 : (lev == 2) ? l2 : l3;
  int wl = w >> lev, hl = h >> lev;
  const float* img = base + (size_t)pix * wl * hl;
  __builtin_prefetch(img, 0, 1);  // global_prefetch_b8: corr tile lives in L2
  float inv = 1.0f / (float)(1 << lev);
  float gx = cx * inv + d0;
  float gy = cy * inv + d1;
  float xs = (gx + 1.0f) * 0.5f * (float)(wl - 1);
  float ys = (gy + 1.0f) * 0.5f * (float)(hl - 1);
  float xf = floorf(xs), yf = floorf(ys);
  float wx = xs - xf, wy = ys - yf;
  int X0 = (int)xf, Y0 = (int)yf;
  auto g = [&](int X, int Y) -> float {
    if (X < 0 || X >= wl || Y < 0 || Y >= hl) return 0.0f;
    return img[(size_t)Y * wl + X];
  };
  float v = g(X0, Y0) * (1 - wx) * (1 - wy) + g(X0 + 1, Y0) * wx * (1 - wy) +
            g(X0, Y0 + 1) * (1 - wx) * wy + g(X0 + 1, Y0 + 1) * wx * wy;
  out[(size_t)(lev * 81 + tap) * HW + pix] = v;
}

// ---------------------------------------------------------------------------
// Bilinear resize (half-pixel centers, edge clamp) with optional scale
// ---------------------------------------------------------------------------
__global__ void k_resize_bilinear(const float* __restrict__ in, float* __restrict__ out,
                                  int C, int ih, int iw, int oh, int ow, float mul) {
  long total = (long)C * oh * ow;
  long i = (long)blockIdx.x * blockDim.x + threadIdx.x;
  if (i >= total) return;
  int x = (int)(i % ow);
  long t = i / ow;
  int y = (int)(t % oh);
  int c = (int)(t / oh);
  float sx = ((x + 0.5f) * iw) / ow - 0.5f;
  float sy = ((y + 0.5f) * ih) / oh - 0.5f;
  sx = fminf(fmaxf(sx, 0.0f), (float)(iw - 1));
  sy = fminf(fmaxf(sy, 0.0f), (float)(ih - 1));
  int x0 = (int)sx, y0 = (int)sy;
  int x1 = (x0 + 1 < iw) ? x0 + 1 : iw - 1;
  int y1 = (y0 + 1 < ih) ? y0 + 1 : ih - 1;
  float fx = sx - x0, fy = sy - y0;
  const float* p = in + (size_t)c * ih * iw;
  float v = p[(size_t)y0 * iw + x0] * (1 - fx) * (1 - fy) +
            p[(size_t)y0 * iw + x1] * fx * (1 - fy) +
            p[(size_t)y1 * iw + x0] * (1 - fx) * fy +
            p[(size_t)y1 * iw + x1] * fx * fy;
  out[i] = v * mul;
}

// ============================================================================
// Host orchestration
// ============================================================================
namespace {

struct Arena {
  char* base;
  size_t off, cap;
  void* alloc(size_t n) {
    off = (off + 255) & ~(size_t)255;
    void* r = base + off;
    off += n;
    return r;
  }
};

inline int cdiv(long a, int b) { return (int)((a + b - 1) / b); }

void gemm_launch(hipStream_t st, const __bf16* A, const __bf16* B, const float* bias,
                 float* C, int M, int N, int K, int tA, int tB, int tC, int act,
                 float scale, int use_tdm) {
  dim3 g(cdiv(N, GEMM_TN), cdiv(M, GEMM_TM));
  k_gemm<<<g, 256, 0, st>>>(A, B, bias, C, M, N, K, tA, tB, tC, act, scale, use_tdm);
}

// conv2d (NCHW in -> NCHW out) as im2col + WMMA GEMM. act: 0 none, 1 relu.
void convh(hipStream_t st, char* scr, size_t scrcap, const float* x, int C, int H,
           int W, const float* wgt, const float* bias, int Cout, int KH, int KW,
           int SH, int SW, int PH, int PW, int act, float* out) {
  int OH = (H + 2 * PH - KH) / SH + 1;
  int OW = (W + 2 * PW - KW) / SW + 1;
  long M = (long)OH * OW;
  long K = (long)C * KH * KW;
  Arena a{scr, 0, scrcap};
  __bf16* col = (__bf16*)a.alloc((size_t)(M * K) * sizeof(__bf16));
  __bf16* wbf = (__bf16*)a.alloc((size_t)Cout * K * sizeof(__bf16));
  long tot = M * K;
  k_im2col<<<cdiv(tot, 256), 256, 0, st>>>(x, col, C, H, W, KH, KW, SH, SW, PH, PW, OH, OW);
  long wn = (long)Cout * K;
  k_f32_to_bf16<<<cdiv(wn, 256), 256, 0, st>>>(wgt, wbf, wn);
  // weights are OIHW (Cout x K) -> transB; transC -> NCHW output
  gemm_launch(st, col, wbf, bias, out, (int)M, Cout, (int)K, 0, 1, 1, act, 1.0f, 0);
}

void d2d(hipStream_t st, void* dst, const void* src, size_t bytes) {
  (void)hipMemcpyAsync(dst, src, bytes, hipMemcpyDeviceToDevice, st);
}

}  // namespace

extern "C" void kernel_launch(void* const* d_in, const int* in_sizes, int n_in,
                              void* d_out, int out_size, void* d_ws, size_t ws_size,
                              hipStream_t stream) {
  (void)in_sizes; (void)n_in; (void)out_size;
  // Input index map (setup_inputs insertion order): image1, image2, then params
  // fnet(w1,b1..w4,b4) cnet(..) menc(c1w..ob) gru(z1w..q2b) fhead init unc, iters.
  enum {
    IN_IMG1 = 0, IN_IMG2 = 1,
    P_FNET = 2, P_CNET = 10, P_MENC = 18, P_GRU = 28,
    P_FHEAD = 40, P_INIT = 44, P_UNC = 48, IN_ITERS = 52
  };
#define PF(i) ((const float*)d_in[(i)])
  const int h = 64, w = 64, HW = 4096;
  const int ITERS = 12;  // fixed per reference setup (graph-capture-safe)

  // carve workspace: tail region is conv scratch (im2col + bf16 weights)
  const size_t SCRATCH = (size_t)64 << 20;
  size_t pcap = (ws_size > SCRATCH) ? ws_size - SCRATCH : ws_size / 2;
  Arena P{(char*)d_ws, 0, pcap};
  char* scr = (char*)d_ws + pcap;
  size_t scrcap = ws_size - pcap;

  // ---- persistent buffers ----
  float* fmap1 = (float*)P.alloc(256l * HW * 4);
  float* fmap2 = (float*)P.alloc(256l * HW * 4);
  float* cout_ = (float*)P.alloc(256l * HW * 4);
  float* net   = (float*)P.alloc(128l * HW * 4);
  float* inp   = (float*)P.alloc(128l * HW * 4);
  float* corr0 = (float*)P.alloc((size_t)HW * HW * 4);        // 64 MB (L2-resident)
  float* corr1 = (float*)P.alloc((size_t)HW * 32 * 32 * 4);   // 16 MB
  float* corr2 = (float*)P.alloc((size_t)HW * 16 * 16 * 4);
  float* corr3 = (float*)P.alloc((size_t)HW * 8 * 8 * 4);
  float* coords0 = (float*)P.alloc(2l * HW * 4);
  float* coords1 = (float*)P.alloc(2l * HW * 4);

  // ---- encoders (fnet on img1, img2; cnet on img1) ----
  auto encoder = [&](const float* img, int base, int c2, int c3, float* fmap) {
    size_t mk = P.off;
    float* t1 = (float*)P.alloc(64l * 256 * 256 * 4);
    convh(stream, scr, scrcap, img, 3, 512, 512, PF(base + 0), PF(base + 1),
          64, 7, 7, 2, 2, 3, 3, 0, t1);
    k_instnorm_relu<<<64, 256, 0, stream>>>(t1, 256 * 256);
    float* t2 = (float*)P.alloc((long)c2 * 128 * 128 * 4);
    convh(stream, scr, scrcap, t1, 64, 256, 256, PF(base + 2), PF(base + 3),
          c2, 3, 3, 2, 2, 1, 1, 0, t2);
    k_instnorm_relu<<<c2, 256, 0, stream>>>(t2, 128 * 128);
    float* t3 = (float*)P.alloc((long)c3 * 64 * 64 * 4);
    convh(stream, scr, scrcap, t2, c2, 128, 128, PF(base + 4), PF(base + 5),
          c3, 3, 3, 2, 2, 1, 1, 0, t3);
    k_instnorm_relu<<<c3, 256, 0, stream>>>(t3, 64 * 64);
    convh(stream, scr, scrcap, t3, c3, 64, 64, PF(base + 6), PF(base + 7),
          256, 1, 1, 1, 1, 0, 0, 1, fmap);
    P.off = mk;
  };
  encoder(PF(IN_IMG1), P_FNET, 96, 128, fmap1);
  encoder(PF(IN_IMG2), P_FNET, 96, 128, fmap2);
  encoder(PF(IN_IMG1), P_CNET, 128, 256, cout_);
  k_split_tanh_relu<<<cdiv(128l * HW, 256), 256, 0, stream>>>(cout_, net, inp, HW);

  // ---- correlation volume: (HW x C) * (C x HW), TDM-staged B tiles ----
  {
    Arena a{scr, 0, scrcap};
    __bf16* f1b = (__bf16*)a.alloc(256l * HW * sizeof(__bf16));
    __bf16* f2b = (__bf16*)a.alloc(256l * HW * sizeof(__bf16));
    k_f32_to_bf16<<<cdiv(256l * HW, 256), 256, 0, stream>>>(fmap1, f1b, 256l * HW);
    k_f32_to_bf16<<<cdiv(256l * HW, 256), 256, 0, stream>>>(fmap2, f2b, 256l * HW);
    // A = fmap1 stored CxHW -> transA; B = fmap2 CxHW = KxN -> TDM path
    gemm_launch(stream, f1b, f2b, nullptr, corr0, HW, HW, 256,
                1, 0, 0, 0, 0.0625f /*1/sqrt(256)*/, 1 /*use_tdm*/);
  }
  k_pool2x2<<<cdiv((long)HW * 32 * 32, 256), 256, 0, stream>>>(corr0, corr1, HW, 64, 64);
  k_pool2x2<<<cdiv((long)HW * 16 * 16, 256), 256, 0, stream>>>(corr1, corr2, HW, 32, 32);
  k_pool2x2<<<cdiv((long)HW * 8 * 8, 256), 256, 0, stream>>>(corr2, corr3, HW, 16, 16);

  // ---- flow init head ----
  {
    size_t mk = P.off;
    float* ih = (float*)P.alloc(128l * HW * 4);
    convh(stream, scr, scrcap, cout_, 256, h, w, PF(P_INIT + 0), PF(P_INIT + 1),
          128, 3, 3, 1, 1, 1, 1, 1, ih);
    float* fi = (float*)P.alloc(2l * HW * 4);
    convh(stream, scr, scrcap, ih, 128, h, w, PF(P_INIT + 2), PF(P_INIT + 3),
          2, 3, 3, 1, 1, 1, 1, 0, fi);
    k_init_coords<<<cdiv(2l * HW, 256), 256, 0, stream>>>(coords0, coords1, fi, h, w);
    P.off = mk;
  }

  // ---- GRU half step helper ----
  auto gru_half = [&](const float* x, int zi, int ri, int qi, int KH, int KW,
                      int PH, int PW) {
    size_t mk = P.off;
    float* hx = (float*)P.alloc(384l * HW * 4);
    d2d(stream, hx, net, 128l * HW * 4);
    d2d(stream, hx + 128l * HW, x, 256l * HW * 4);
    float* zpre = (float*)P.alloc(128l * HW * 4);
    convh(stream, scr, scrcap, hx, 384, h, w, PF(zi), PF(zi + 1), 128, KH, KW, 1, 1, PH, PW, 0, zpre);
    float* rpre = (float*)P.alloc(128l * HW * 4);
    convh(stream, scr, scrcap, hx, 384, h, w, PF(ri), PF(ri + 1), 128, KH, KW, 1, 1, PH, PW, 0, rpre);
    float* z  = (float*)P.alloc(128l * HW * 4);
    float* rh = (float*)P.alloc(128l * HW * 4);
    k_sigmoid_mul<<<cdiv(128l * HW, 256), 256, 0, stream>>>(zpre, rpre, net, z, rh, 128l * HW);
    float* qin = (float*)P.alloc(384l * HW * 4);
    d2d(stream, qin, rh, 128l * HW * 4);
    d2d(stream, qin + 128l * HW, x, 256l * HW * 4);
    float* qpre = (float*)P.alloc(128l * HW * 4);
    convh(stream, scr, scrcap, qin, 384, h, w, PF(qi), PF(qi + 1), 128, KH, KW, 1, 1, PH, PW, 0, qpre);
    k_gru_out<<<cdiv(128l * HW, 256), 256, 0, stream>>>(z, qpre, net, 128l * HW);
    P.off = mk;
  };

  // ---- iterative updates ----
  for (int it = 0; it < ITERS; ++it) {
    size_t mk = P.off;
    float* cfeat = (float*)P.alloc(324l * HW * 4);
    k_corr_lookup<<<cdiv((long)HW * 4 * 81, 256), 256, 0, stream>>>(
        corr0, corr1, corr2, corr3, coords1, cfeat, h, w);
    float* flow = (float*)P.alloc(2l * HW * 4);
    k_sub<<<cdiv(2l * HW, 256), 256, 0, stream>>>(coords1, coords0, flow, 2l * HW);
    // motion encoder
    float* c1 = (float*)P.alloc(256l * HW * 4);
    convh(stream, scr, scrcap, cfeat, 324, h, w, PF(P_MENC + 0), PF(P_MENC + 1), 256, 1, 1, 1, 1, 0, 0, 1, c1);
    float* c2o = (float*)P.alloc(192l * HW * 4);
    convh(stream, scr, scrcap, c1, 256, h, w, PF(P_MENC + 2), PF(P_MENC + 3), 192, 3, 3, 1, 1, 1, 1, 1, c2o);
    float* f1 = (float*)P.alloc(128l * HW * 4);
    convh(stream, scr, scrcap, flow, 2, h, w, PF(P_MENC + 4), PF(P_MENC + 5), 128, 7, 7, 1, 1, 3, 3, 1, f1);
    float* f2 = (float*)P.alloc(64l * HW * 4);
    convh(stream, scr, scrcap, f1, 128, h, w, PF(P_MENC + 6), PF(P_MENC + 7), 64, 3, 3, 1, 1, 1, 1, 1, f2);
    float* cat = (float*)P.alloc(256l * HW * 4);
    d2d(stream, cat, c2o, 192l * HW * 4);
    d2d(stream, cat + 192l * HW, f2, 64l * HW * 4);
    float* owo = (float*)P.alloc(126l * HW * 4);
    convh(stream, scr, scrcap, cat, 256, h, w, PF(P_MENC + 8), PF(P_MENC + 9), 126, 3, 3, 1, 1, 1, 1, 1, owo);
    float* x = (float*)P.alloc(256l * HW * 4);  // x = [inp | mf]; mf = [owo | flow]
    d2d(stream, x, inp, 128l * HW * 4);
    d2d(stream, x + 128l * HW, owo, 126l * HW * 4);
    d2d(stream, x + 254l * HW, flow, 2l * HW * 4);
    // sep conv GRU
    gru_half(x, P_GRU + 0, P_GRU + 2, P_GRU + 4, 1, 5, 0, 2);
    gru_half(x, P_GRU + 6, P_GRU + 8, P_GRU + 10, 5, 1, 2, 0);
    // flow head
    float* fh = (float*)P.alloc(256l * HW * 4);
    convh(stream, scr, scrcap, net, 128, h, w, PF(P_FHEAD + 0), PF(P_FHEAD + 1), 256, 3, 3, 1, 1, 1, 1, 1, fh);
    float* delta = (float*)P.alloc(2l * HW * 4);
    convh(stream, scr, scrcap, fh, 256, h, w, PF(P_FHEAD + 2), PF(P_FHEAD + 3), 2, 3, 3, 1, 1, 1, 1, 0, delta);
    k_add_inplace<<<cdiv(2l * HW, 256), 256, 0, stream>>>(coords1, delta, 2l * HW);
    P.off = mk;
  }

  // ---- outputs: flow_up (2,512,512) then unc_up (1,512,512) ----
  {
    size_t mk = P.off;
    float* flow = (float*)P.alloc(2l * HW * 4);
    k_sub<<<cdiv(2l * HW, 256), 256, 0, stream>>>(coords1, coords0, flow, 2l * HW);
    k_resize_bilinear<<<cdiv(2l * 512 * 512, 256), 256, 0, stream>>>(
        flow, (float*)d_out, 2, h, w, 512, 512, 8.0f);
    float* u1 = (float*)P.alloc(64l * HW * 4);
    convh(stream, scr, scrcap, net, 128, h, w, PF(P_UNC + 0), PF(P_UNC + 1), 64, 3, 3, 1, 1, 1, 1, 1, u1);
    float* u2 = (float*)P.alloc(1l * HW * 4);
    convh(stream, scr, scrcap, u1, 64, h, w, PF(P_UNC + 2), PF(P_UNC + 3), 1, 3, 3, 1, 1, 1, 1, 0, u2);
    k_softplus<<<cdiv((long)HW, 256), 256, 0, stream>>>(u2, HW);
    k_resize_bilinear<<<cdiv(1l * 512 * 512, 256), 256, 0, stream>>>(
        u2, (float*)d_out + 2l * 512 * 512, 1, h, w, 512, 512, 1.0f);
    P.off = mk;
  }
#undef PF
}
